// Actor_7842610283028
// MI455X (gfx1250) — matrix operand order repaired and checked by reading, not compiled, not run on hardware
//
#include <hip/hip_runtime.h>
#include <hip/hip_bf16.h>
#include <stdint.h>

// ---------------- problem constants (from reference) ----------------
#define BATCH     4096
#define STATE     1024
#define HID       256
#define ACT       256
#define MAX_ITERS 50
#define TOL       1e-4

// ---------------- CDNA5 WMMA plumbing ----------------
typedef __attribute__((ext_vector_type(16))) __bf16      v16bf;
typedef __attribute__((ext_vector_type(8)))  float       v8f;
typedef __attribute__((ext_vector_type(4)))  unsigned int v4u;

// trivial ext-vector members -> union is legal (HIP's uint4 has ctors)
union Frag { v4u q[2]; v16bf v; };

__device__ __forceinline__ unsigned short bf16_rne(float f) {
  unsigned int u = __float_as_uint(f);
  u += 0x7fffu + ((u >> 16) & 1u);          // round-to-nearest-even
  return (unsigned short)(u >> 16);
}
__device__ __forceinline__ float bf16_f32(unsigned short h) {
  return __uint_as_float(((unsigned int)h) << 16);
}
__device__ __forceinline__ v8f wmma_bf16(v16bf a, v16bf b, v8f c) {
  // 8-arg form: (neg_a, A, neg_b, B, c_mod, C, reuse_a, reuse_b)
  return __builtin_amdgcn_wmma_f32_16x16x32_bf16(false, a, false, b, (short)0, c,
                                                 false, false);
}

// gfx1250 has a hardware V_TANH_F32 (TRANS32 op); use it when exposed.
#if __has_builtin(__builtin_amdgcn_tanhf)
__device__ __forceinline__ float fast_tanh(float x) { return __builtin_amdgcn_tanhf(x); }
#else
__device__ __forceinline__ float fast_tanh(float x) { return tanhf(x); }
#endif

// ---------------- weight split: f32 -> bf16 hi + bf16 lo ----------------
__global__ void split_bf16(const float* __restrict__ w,
                           unsigned short* __restrict__ hi,
                           unsigned short* __restrict__ lo, int n) {
  int i = blockIdx.x * 256 + threadIdx.x;
  if (i < n) {
    float x = w[i];
    unsigned short h = bf16_rne(x);
    hi[i] = h;
    lo[i] = bf16_rne(x - bf16_f32(h));
  }
}

// ---------------- fixed-point bookkeeping ----------------
__global__ void fp_init(double* err2, int* done, int* cur) {
  *err2 = 0.0; *done = 0; *cur = 0;
}
// Frobenius check matches the reference: done |= (||z - z_next|| < TOL).
// `cur` tracks which ping-pong buffer holds the live z.
__global__ void fp_check(double* err2, int* done, int* cur) {
  if (*done) return;
  *cur ^= 1;
  if (*err2 < (double)TOL * (double)TOL) *done = 1;
  *err2 = 0.0;
}

// ---------------- bf16x3 WMMA GEMM with fused epilogues ----------------
// C[M,N] = epi( A[M,K] @ W^T + ... ), W row-major [N,K] pre-split to bf16
// hi/lo. B = W^T means K-contiguous columns of B are rows of W -> both LDS
// tiles are row-major-in-K and fragments are plain ds_load_b128 pairs.
// Software pipelined: next K-chunk is prefetched into VGPRs while the
// current chunk's 24 WMMAs execute.
//
// MODE 0: out = tanh(acc + bias[n]);  out2 = tanh(out)      (layer1 + z1 seed)
// MODE 1: out = tanh(acc + addmat);   err2 += ||out - A||^2 (fixed-point step)
// MODE 2: out = tanh(acc + bias[n])   (A selected by *sel between A0/A1)
// MODE 3: out = tanh(acc + bias[n]) * scale
template <int MODE>
__global__ __launch_bounds__(256)
void gemm_bf16x3(const float* __restrict__ A0, const float* __restrict__ A1,
                 const int* __restrict__ sel,
                 const unsigned short* __restrict__ Bhi,
                 const unsigned short* __restrict__ Blo,
                 const float* __restrict__ bias,
                 const float* __restrict__ addmat,
                 float* __restrict__ out, float* __restrict__ out2,
                 int M, int N, int K, float scale,
                 const int* __restrict__ done, double* __restrict__ err2) {
  if (MODE == 1) {
    if (*done) return;  // converged: freeze z (uniform exit, matches reference)
  }
  const float* A = A0;
  if (MODE == 2) { if (sel && *sel) A = A1; }

  // 128x32 tiles, rows padded to 40 bf16 (80B): the 16 fragment lanes'
  // ds_load_b128s hit disjoint bank groups.
  __shared__ __align__(16) unsigned short sAh[128][40];
  __shared__ __align__(16) unsigned short sAl[128][40];
  __shared__ __align__(16) unsigned short sBh[128][40];
  __shared__ __align__(16) unsigned short sBl[128][40];
  __shared__ float wred[8];

  const int tid  = threadIdx.x;
  const int m0   = blockIdx.x * 128;
  const int n0   = blockIdx.y * 128;
  const int wave = tid >> 5;       // 8 waves: 4 along M x 2 along N
  const int lane = tid & 31;
  const int ln   = lane & 15;
  const int grp  = lane >> 4;
  const int wm   = wave & 3;       // 32 rows per wave  (2 M-tiles)
  const int wn   = wave >> 2;      // 64 cols per wave  (4 N-tiles)

  // staging coordinates (per thread, constant across K-chunks)
  const int ar = tid >> 3;          // A row (+32*i), col (tid&7)*4
  const int ac = (tid & 7) << 2;
  const int br = tid >> 2;          // B row (+64*i), col (tid&3)*8
  const int bc = (tid & 3) << 3;
  const float*          pA  = A   + (size_t)(m0 + ar) * K + ac;
  const unsigned short* pBh = Bhi + (size_t)(n0 + br) * K + bc;
  const unsigned short* pBl = Blo + (size_t)(n0 + br) * K + bc;

  float4 ra[4];          // prefetched A chunk (f32)
  v4u    rbh[2], rbl[2]; // prefetched B chunk (bf16 hi/lo)

  auto load_chunk = [&](int kc) {
#pragma unroll
    for (int i = 0; i < 4; ++i)
      ra[i] = *(const float4*)(pA + (size_t)(32 * i) * K + kc);
#pragma unroll
    for (int i = 0; i < 2; ++i) {
      rbh[i] = *(const v4u*)(pBh + (size_t)(64 * i) * K + kc);
      rbl[i] = *(const v4u*)(pBl + (size_t)(64 * i) * K + kc);
    }
  };
  auto store_chunk = [&]() {
#pragma unroll
    for (int i = 0; i < 4; ++i) {
      const float4 a = ra[i];
      unsigned short h0 = bf16_rne(a.x), h1 = bf16_rne(a.y);
      unsigned short h2 = bf16_rne(a.z), h3 = bf16_rne(a.w);
      unsigned short l0 = bf16_rne(a.x - bf16_f32(h0));
      unsigned short l1 = bf16_rne(a.y - bf16_f32(h1));
      unsigned short l2 = bf16_rne(a.z - bf16_f32(h2));
      unsigned short l3 = bf16_rne(a.w - bf16_f32(h3));
      uint2 ph, pl;
      ph.x = (unsigned)h0 | ((unsigned)h1 << 16);
      ph.y = (unsigned)h2 | ((unsigned)h3 << 16);
      pl.x = (unsigned)l0 | ((unsigned)l1 << 16);
      pl.y = (unsigned)l2 | ((unsigned)l3 << 16);
      *(uint2*)&sAh[ar + 32 * i][ac] = ph;
      *(uint2*)&sAl[ar + 32 * i][ac] = pl;
    }
#pragma unroll
    for (int i = 0; i < 2; ++i) {
      *(v4u*)&sBh[br + 64 * i][bc] = rbh[i];
      *(v4u*)&sBl[br + 64 * i][bc] = rbl[i];
    }
  };

  v8f acc[2][4] = {};

  load_chunk(0);  // prologue prefetch
  for (int kc = 0; kc < K; kc += 32) {
    __syncthreads();          // previous compute done reading LDS
    store_chunk();            // split + commit prefetched chunk
    __syncthreads();
    if (kc + 32 < K) load_chunk(kc + 32);  // in flight under the WMMAs

    // --- fragments (CDNA5 16-bit A 16x32 / B 32x16 VGPR layouts) ---
    Frag ah[2], al[2], bh[4], bl[4];
#pragma unroll
    for (int i = 0; i < 2; ++i) {
      const unsigned short* ph = &sAh[wm * 32 + i * 16 + ln][grp * 8];
      ah[i].q[0] = *(const v4u*)(ph);          // K = 8*grp .. +7
      ah[i].q[1] = *(const v4u*)(ph + 16);     // K = 8*grp+16 .. +23
      const unsigned short* pl = &sAl[wm * 32 + i * 16 + ln][grp * 8];
      al[i].q[0] = *(const v4u*)(pl);
      al[i].q[1] = *(const v4u*)(pl + 16);
    }
#pragma unroll
    for (int j = 0; j < 4; ++j) {
      const unsigned short* ph = &sBh[wn * 64 + j * 16 + ln][grp * 8];
      bh[j].q[0] = *(const v4u*)(ph);          // K = 8*grp .. +7
      bh[j].q[1] = *(const v4u*)(ph + 16);     // K = 8*grp+16 .. +23
      const unsigned short* pl = &sBl[wn * 64 + j * 16 + ln][grp * 8];
      bl[j].q[0] = *(const v4u*)(pl);
      bl[j].q[1] = *(const v4u*)(pl + 16);
    }

    // bf16x3: smallest terms first, all into the same f32 accumulator
#pragma unroll
    for (int i = 0; i < 2; ++i)
#pragma unroll
      for (int j = 0; j < 4; ++j) {
        acc[i][j] = wmma_bf16(al[i].v, bh[j].v, acc[i][j]);
        acc[i][j] = wmma_bf16(ah[i].v, bl[j].v, acc[i][j]);
        acc[i][j] = wmma_bf16(ah[i].v, bh[j].v, acc[i][j]);
      }
  }

  // --- epilogue (C/D layout: vgpr v -> row grp*8+v, col = ln) ---
  float lsum = 0.f;
#pragma unroll
  for (int j = 0; j < 4; ++j) {
    const int col = n0 + wn * 64 + j * 16 + ln;
    float bv = 0.f;
    if (MODE != 1) bv = bias[col];
#pragma unroll
    for (int i = 0; i < 2; ++i) {
      const int row0 = m0 + wm * 32 + i * 16 + grp * 8;
#pragma unroll
      for (int v = 0; v < 8; ++v) {
        const size_t rc = (size_t)(row0 + v) * N + col;
        float s = acc[i][j][v];
        if (MODE == 0) {
          s = fast_tanh(s + bv);
          out[rc]  = s;              // x_in for the fixed point
          out2[rc] = fast_tanh(s);   // z1 = tanh(0 @ W_fp + x_in)
        } else if (MODE == 1) {
          s = fast_tanh(s + addmat[rc]);
          const float d = s - A[rc];     // A == z_old
          lsum = fmaf(d, d, lsum);
          out[rc] = s;
        } else if (MODE == 2) {
          out[rc] = fast_tanh(s + bv);
        } else {
          out[rc] = fast_tanh(s + bv) * scale;
        }
      }
    }
  }

  if (MODE == 1) {  // block-reduce ||dz||^2, one f64 atomic per block
#pragma unroll
    for (int off = 16; off > 0; off >>= 1) lsum += __shfl_down(lsum, off);
    if (lane == 0) wred[wave] = lsum;
    __syncthreads();
    if (tid == 0) {
      double t = 0.0;
#pragma unroll
      for (int k = 0; k < 8; ++k) t += (double)wred[k];
      atomicAdd(err2, t);
    }
  }
}

// ---------------- host orchestration ----------------
extern "C" void kernel_launch(void* const* d_in, const int* in_sizes, int n_in,
                              void* d_out, int out_size, void* d_ws, size_t ws_size,
                              hipStream_t stream) {
  (void)in_sizes; (void)n_in; (void)out_size; (void)ws_size;
  const float* x    = (const float*)d_in[0];
  const float* W_t  = (const float*)d_in[1];
  const float* b_t  = (const float*)d_in[2];
  const float* W_fp = (const float*)d_in[3];
  const float* W_h  = (const float*)d_in[4];
  const float* b_h  = (const float*)d_in[5];
  const float* W_o  = (const float*)d_in[6];
  const float* b_o  = (const float*)d_in[7];
  float* out = (float*)d_out;

  // workspace carve (~60 MB total; entire working set is L2-resident)
  char*  base = (char*)d_ws;
  size_t off  = 0;
  auto carve = [&](size_t bytes) -> char* {
    char* r = base + off;
    off = (off + bytes + 255) & ~(size_t)255;
    return r;
  };
  float* xin  = (float*)carve((size_t)BATCH * STATE * 4);
  float* zA   = (float*)carve((size_t)BATCH * STATE * 4);
  float* zB   = (float*)carve((size_t)BATCH * STATE * 4);
  float* hbuf = (float*)carve((size_t)BATCH * HID * 4);
  unsigned short* wt_hi  = (unsigned short*)carve((size_t)STATE * STATE * 2);
  unsigned short* wt_lo  = (unsigned short*)carve((size_t)STATE * STATE * 2);
  unsigned short* wfp_hi = (unsigned short*)carve((size_t)STATE * STATE * 2);
  unsigned short* wfp_lo = (unsigned short*)carve((size_t)STATE * STATE * 2);
  unsigned short* wh_hi  = (unsigned short*)carve((size_t)HID * STATE * 2);
  unsigned short* wh_lo  = (unsigned short*)carve((size_t)HID * STATE * 2);
  unsigned short* wo_hi  = (unsigned short*)carve((size_t)ACT * HID * 2);
  unsigned short* wo_lo  = (unsigned short*)carve((size_t)ACT * HID * 2);
  double* err2 = (double*)carve(sizeof(double));
  int*    done = (int*)carve(sizeof(int));
  int*    cur  = (int*)carve(sizeof(int));

  fp_init<<<dim3(1), dim3(1), 0, stream>>>(err2, done, cur);

  split_bf16<<<dim3((STATE * STATE + 255) / 256), dim3(256), 0, stream>>>(
      W_t, wt_hi, wt_lo, STATE * STATE);
  split_bf16<<<dim3((STATE * STATE + 255) / 256), dim3(256), 0, stream>>>(
      W_fp, wfp_hi, wfp_lo, STATE * STATE);
  split_bf16<<<dim3((HID * STATE + 255) / 256), dim3(256), 0, stream>>>(
      W_h, wh_hi, wh_lo, HID * STATE);
  split_bf16<<<dim3((ACT * HID + 255) / 256), dim3(256), 0, stream>>>(
      W_o, wo_hi, wo_lo, ACT * HID);

  const dim3 blk(256);
  const dim3 gFull(BATCH / 128, STATE / 128);   // 32 x 8

  // Layer 1: x_in = tanh(x @ W_t^T + b_t); also seed z1 = tanh(x_in)
  gemm_bf16x3<0><<<gFull, blk, 0, stream>>>(x, nullptr, nullptr, wt_hi, wt_lo,
                                            b_t, nullptr, xin, zA,
                                            BATCH, STATE, STATE, 1.f,
                                            nullptr, nullptr);

  // Fixed point: 49 remaining steps (step 1 was the z1 seed); converged
  // iterations early-exit on the device `done` flag -> exact freeze semantics.
  for (int it = 0; it < MAX_ITERS - 1; ++it) {
    float* src = (it & 1) ? zB : zA;
    float* dst = (it & 1) ? zA : zB;
    gemm_bf16x3<1><<<gFull, blk, 0, stream>>>(src, nullptr, nullptr, wfp_hi,
                                              wfp_lo, nullptr, xin, dst, nullptr,
                                              BATCH, STATE, STATE, 1.f,
                                              done, err2);
    fp_check<<<dim3(1), dim3(1), 0, stream>>>(err2, done, cur);
  }

  // Hidden layer: picks the live z buffer via *cur at runtime
  const dim3 gHid(BATCH / 128, HID / 128);      // 32 x 2
  gemm_bf16x3<2><<<gHid, blk, 0, stream>>>(zA, zB, cur, wh_hi, wh_lo, b_h,
                                           nullptr, hbuf, nullptr,
                                           BATCH, HID, STATE, 1.f,
                                           nullptr, nullptr);

  // Output layer: tanh(...) * ACT
  const dim3 gOut(BATCH / 128, ACT / 128);      // 32 x 2
  gemm_bf16x3<3><<<gOut, blk, 0, stream>>>(hbuf, nullptr, nullptr, wo_hi, wo_lo,
                                           b_o, nullptr, out, nullptr,
                                           BATCH, ACT, HID, (float)ACT,
                                           nullptr, nullptr);
}
